// SparseCostReg_40570261078322
// MI455X (gfx1250) — matrix operand order repaired and checked by reading, not compiled, not run on hardware
//
#include <hip/hip_runtime.h>

// ---------------------------------------------------------------------------
// SparseCostReg 3D U-Net on gfx1250 via bf16 WMMA implicit GEMM.
// Activations stored CHANNEL-LAST ([voxel][C]) so every A-fragment group
// (one tap x 8 consecutive channels) is two global_load_b128's; out-of-bounds
// groups read from a zero page (pointer select, no data masking).
// All convs / transpose convs lower onto v_wmma_f32_16x16x32_bf16.
// ---------------------------------------------------------------------------

typedef __attribute__((ext_vector_type(16))) __bf16 v16bf;
typedef __attribute__((ext_vector_type(4)))  __bf16 v4bf;
typedef __attribute__((ext_vector_type(8)))  float  v8f;
typedef __attribute__((ext_vector_type(4)))  float  v4f;
typedef __attribute__((ext_vector_type(8)))  int    v8i;

struct TapPack {
  int t[32];   // (kd<<16)|(kh<<8)|kw per tap (zero padded)
  int o[32];   // precomputed ((kd*Hi+kh)*Wi+kw)*Cin per tap
};

// ---------------------------------------------------------------------------
// Generic direct conv, fused affine+ReLU+residual, channel-last activations.
// Block = 256 threads = 8 waves; each wave owns TWO 16-voxel M tiles sharing
// one B fragment; 16-Cout bf16 weight panel staged pair-packed in LDS.
// K is flattened tap-major, channel-minor (Cin power of two -> shift decode).
// ---------------------------------------------------------------------------
__global__ __launch_bounds__(256) void conv3d_wmma_kernel(
    const float* __restrict__ x, const float* __restrict__ w,
    const float* __restrict__ scale, const float* __restrict__ shift,
    const float* __restrict__ resid, float* __restrict__ y,
    const float* __restrict__ zpad,
    int Cin, int cinShift, int Cout,
    int Di, int Hi, int Wi, int Do, int Ho, int Wo,
    int Ktap, int Ktot, int KtotP,
    int stride, int pd, int ph, int pw, int relu, TapPack tp)
{
  extern __shared__ unsigned char smem[];
  int*    tapLds  = (int*)smem;          // 32 packed tap coords
  int*    toffLds = (int*)(smem + 128);  // 32 tap offsets (pre-scaled by Cin)
  __bf16* wlds    = (__bf16*)(smem + 256);

  const int tid = threadIdx.x;
  if (tid < 32) { tapLds[tid] = tp.t[tid]; toffLds[tid] = tp.o[tid]; }

  const int coutBase = blockIdx.y << 4;
  const int Vout = Do * Ho * Wo;

  // Weight panel, pair-packed along K for dword B-fragment reads:
  // element (k, co) lives at bf16 index (k>>1)*32 + co*2 + (k&1)
  for (int idx = tid; idx < (KtotP << 4); idx += 256) {
    int co  = idx / KtotP;
    int k   = idx - co * KtotP;
    int cog = coutBase + co;
    float v = 0.f;
    if (k < Ktot && cog < Cout)
      v = w[cog * Ktot + (k & (Cin - 1)) * Ktap + (k >> cinShift)];
    wlds[((k >> 1) << 5) + (co << 1) + (k & 1)] = (__bf16)v;
  }
  __syncthreads();

  const int lane = tid & 31;
  const int wave = tid >> 5;
  const int half = lane >> 4;
  const int l16  = lane & 15;

  const int ntm = (Vout + 15) >> 4;
  const int t0  = (blockIdx.x * 8 + wave) * 2;   // two M tiles per wave
  if (t0 >= ntm) return;

  bool mval[2]; int spC[2]; int idB[2], ihB[2], iwB[2];
  const int HWo = Ho * Wo;
#pragma unroll
  for (int s = 0; s < 2; ++s) {
    int tM = t0 + s;
    int vm = (tM << 4) + l16;
    bool mv = (tM < ntm) && (vm < Vout);
    mval[s] = mv;
    int vv = mv ? vm : 0;
    int od = vv / HWo;
    int r  = vv - od * HWo;
    int oh = r / Wo;
    int ow = r - oh * Wo;
    idB[s] = od * stride - pd;
    ihB[s] = oh * stride - ph;
    iwB[s] = ow * stride - pw;
    spC[s] = ((idB[s] * Hi + ihB[s]) * Wi + iwB[s]) * Cin;
  }

  // Hoisted per-tap in-bounds masks (one bit per tap per M tile).
  unsigned vmask[2] = {0u, 0u};
  for (int tI = 0; tI < Ktap; ++tI) {
    int tt = tapLds[tI];
    int dz = tt >> 16, dy = (tt >> 8) & 255, dx = tt & 255;
#pragma unroll
    for (int s = 0; s < 2; ++s) {
      int id = idB[s] + dz, ih = ihB[s] + dy, iw = iwB[s] + dx;
      if ((unsigned)id < (unsigned)Di && (unsigned)ih < (unsigned)Hi &&
          (unsigned)iw < (unsigned)Wi)
        vmask[s] |= (1u << tI);
    }
  }

  v8f accs[2] = {};
  const int* wldsI = (const int*)wlds;

  for (int k0 = 0; k0 < KtotP; k0 += 32) {
    // Shared B fragment: 8 dword LDS loads (pairs of bf16 per VGPR).
    v8i braw;
    const int pBase = (k0 + (half << 4)) >> 1;
#pragma unroll
    for (int j = 0; j < 8; ++j)
      braw[j] = wldsI[((pBase + j) << 4) + l16];
    const v16bf b = __builtin_bit_cast(v16bf, braw);

    // Two 8-consecutive-K groups per lane, one tap each; channels contiguous.
    const int kgA = k0 + (half << 4);
    const int kgB = kgA + 8;
    const int tapA = kgA >> cinShift,  tapB = kgB >> cinShift;
    const int ciA  = kgA & (Cin - 1),  ciB  = kgB & (Cin - 1);
    const int toA  = toffLds[tapA],    toB  = toffLds[tapB];
    const bool kokA = kgA < Ktot,      kokB = kgB < Ktot;

#pragma unroll
    for (int s = 0; s < 2; ++s) {
      const bool okA = mval[s] && kokA && ((vmask[s] >> tapA) & 1u);
      const bool okB = mval[s] && kokB && ((vmask[s] >> tapB) & 1u);
      // Pointer select: invalid groups read the zero page.
      const float* pA = okA ? (x + (spC[s] + toA + ciA)) : zpad;
      const float* pB = okB ? (x + (spC[s] + toB + ciB)) : zpad;
      const v4f fA0 = *(const v4f*)pA;
      const v4f fA1 = *(const v4f*)(pA + 4);
      const v4f fB0 = *(const v4f*)pB;
      const v4f fB1 = *(const v4f*)(pB + 4);
      const v4bf cA0 = __builtin_convertvector(fA0, v4bf);
      const v4bf cA1 = __builtin_convertvector(fA1, v4bf);
      const v4bf cB0 = __builtin_convertvector(fB0, v4bf);
      const v4bf cB1 = __builtin_convertvector(fB1, v4bf);
      v16bf a;
#pragma unroll
      for (int i = 0; i < 4; ++i) {
        a[i]      = cA0[i];
        a[i + 4]  = cA1[i];
        a[i + 8]  = cB0[i];
        a[i + 12] = cB1[i];
      }
      accs[s] = __builtin_amdgcn_wmma_f32_16x16x32_bf16(
          false, a, false, b, (short)0, accs[s], false, false);
    }
  }

  // Epilogue per C/D layout: VGPR r, lane -> (M = r + 8*half, N = l16).
  // Channel-last store: lanes 0..15 write 16 consecutive floats.
  const int co = coutBase + l16;
  float sc = 1.f, sh = 0.f;
  if (scale != nullptr && co < Cout) { sc = scale[co]; sh = shift[co]; }
#pragma unroll
  for (int s = 0; s < 2; ++s) {
#pragma unroll
    for (int r = 0; r < 8; ++r) {
      int v = ((t0 + s) << 4) + r + (half << 3);
      if (co < Cout && v < Vout) {
        float o = accs[s][r] * sc + sh;
        if (relu) o = fmaxf(o, 0.f);
        long idx = (long)v * Cout + co;
        if (resid != nullptr) o += resid[idx];
        y[idx] = o;
      }
    }
  }
}

// ---------------------------------------------------------------------------
// Transpose conv 2x2x2, stride 2, VALID: parity-partitioned 1x1x1 conv
// (each output voxel has exactly one contributing tap). Channel-last.
// Block = 256 threads = 8 waves; bf16 weights staged pair-packed per tap.
// LDS layout: [wsub][K/2 pairs][co][2].
// ---------------------------------------------------------------------------
__global__ __launch_bounds__(256) void tconv3d_wmma_kernel(
    const float* __restrict__ x, const float* __restrict__ w,
    const float* __restrict__ scale, const float* __restrict__ shift,
    float* __restrict__ y, const float* __restrict__ zpad,
    int Cin, int Cout, int Di, int Hi, int Wi)
{
  extern __shared__ unsigned char smem[];
  __bf16* wlds = (__bf16*)smem;

  const int tid = threadIdx.x;
  const int KP  = (Cin + 31) & ~31;
  const int coutBase = blockIdx.y << 4;

  for (int idx = tid; idx < (KP << 4) * 8; idx += 256) {
    int ws_ = idx / (KP << 4);
    int rem = idx - ws_ * (KP << 4);
    int co  = rem / KP;
    int k   = rem - co * KP;
    int cog = coutBase + co;
    float v = 0.f;
    if (k < Cin && cog < Cout)
      v = w[((k * Cout + cog) << 3) + ws_];    // w[ci][co][kd][kh][kw]
    wlds[((ws_ * (KP >> 1) + (k >> 1)) << 5) + (co << 1) + (k & 1)] = (__bf16)v;
  }
  __syncthreads();

  const int Do = Di * 2, Ho = Hi * 2, Wo = Wi * 2;
  const int lane = tid & 31, wave = tid >> 5;
  const int half = lane >> 4, l16 = lane & 15;

  const int cpr = (Wi + 15) >> 4;     // 16-output chunks per (d,h,parity) row
  const int nTiles = Do * Ho * 2 * cpr;
  const int tile = blockIdx.x * 8 + wave;
  if (tile >= nTiles) return;

  int bid = tile;
  const int chunk = bid % cpr; bid /= cpr;
  const int par   = bid & 1;   bid >>= 1;
  const int h     = bid % Ho;
  const int d     = bid / Ho;
  const int wsub  = ((d & 1) * 2 + (h & 1)) * 2 + par;  // tap within (2,2,2)
  const int di = d >> 1, hi = h >> 1;

  const int rtC  = ((di * Hi + hi) * Wi + (chunk << 4) + l16) * Cin;
  const bool mvalid = ((chunk << 4) + l16) < Wi;
  const int co  = coutBase + l16;
  const bool cok = co < Cout;

  v8f acc = {};
  const int* wldsI = (const int*)wlds;
  const int tapPairBase = wsub * (KP >> 1);

  for (int k0 = 0; k0 < KP; k0 += 32) {
    // B fragment: 8 dword LDS loads.
    v8i braw;
    const int pBase = tapPairBase + ((k0 + (half << 4)) >> 1);
#pragma unroll
    for (int j = 0; j < 8; ++j)
      braw[j] = wldsI[((pBase + j) << 4) + l16];
    const v16bf b = __builtin_bit_cast(v16bf, braw);

    const int kgA = k0 + (half << 4);
    const int kgB = kgA + 8;
    const bool okA = mvalid && (kgA < Cin);
    const bool okB = mvalid && (kgB < Cin);
    const float* pA = okA ? (x + rtC + kgA) : zpad;
    const float* pB = okB ? (x + rtC + kgB) : zpad;
    const v4f fA0 = *(const v4f*)pA;
    const v4f fA1 = *(const v4f*)(pA + 4);
    const v4f fB0 = *(const v4f*)pB;
    const v4f fB1 = *(const v4f*)(pB + 4);
    const v4bf cA0 = __builtin_convertvector(fA0, v4bf);
    const v4bf cA1 = __builtin_convertvector(fA1, v4bf);
    const v4bf cB0 = __builtin_convertvector(fB0, v4bf);
    const v4bf cB1 = __builtin_convertvector(fB1, v4bf);
    v16bf a;
#pragma unroll
    for (int i = 0; i < 4; ++i) {
      a[i]      = cA0[i];
      a[i + 4]  = cA1[i];
      a[i + 8]  = cB0[i];
      a[i + 12] = cB1[i];
    }
    acc = __builtin_amdgcn_wmma_f32_16x16x32_bf16(false, a, false, b, (short)0,
                                                  acc, false, false);
  }

  float sc = 1.f, sh = 0.f;
  if (scale != nullptr && cok) { sc = scale[co]; sh = shift[co]; }
#pragma unroll
  for (int r = 0; r < 8; ++r) {
    int tR = (chunk << 4) + r + (half << 3);
    if (cok && tR < Wi) {
      int wo = par + (tR << 1);
      int v  = (d * Ho + h) * Wo + wo;
      float o = fmaxf(acc[r] * sc + sh, 0.f);
      y[(long)v * Cout + co] = o;
    }
  }
}

// ---------------------------------------------------------------------------
// Depth scatter to channel-last: x0[(z[v]*H+h)*W+w][c] = cv[c][d][h][w]
// (x0 pre-zeroed)
// ---------------------------------------------------------------------------
__global__ __launch_bounds__(256) void scatter_kernel(
    const float* __restrict__ cv, const int* __restrict__ z,
    float* __restrict__ x0, int D, int H, int W, int C)
{
  long V = (long)D * H * W;
  long v = (long)blockIdx.x * 256 + threadIdx.x;
  if (v >= V) return;
  long HW = (long)H * W;
  int  d  = (int)(v / HW);
  long hw = v - (long)d * HW;
  int  zz = z[v];
  long dst = (long)zz * HW + hw;
  for (int c = 0; c < C; ++c)
    x0[dst * C + c] = cv[(long)c * V + v];
}

// Depth gather: out[d,h,w] = p[z[v], h, w]   (p channel-last with C=1)
__global__ __launch_bounds__(256) void gather_kernel(
    const float* __restrict__ p, const int* __restrict__ z,
    float* __restrict__ out, int D, int H, int W)
{
  long V = (long)D * H * W;
  long v = (long)blockIdx.x * 256 + threadIdx.x;
  if (v >= V) return;
  long HW = (long)H * W;
  long hw = v % HW;
  out[v] = p[(long)z[v] * HW + hw];
}

// ---------------------------------------------------------------------------
// Host side
// ---------------------------------------------------------------------------
struct Dims { int D, H, W; };

static void launch_conv(hipStream_t s,
    const float* x, const float* w, const float* scale, const float* shift,
    const float* resid, float* y, const float* zpad,
    int Cin, int Cout, Dims in, Dims out,
    int KD, int KH, int KW, int stride, int pd, int ph, int pw, int relu)
{
  int Ktap = KD * KH * KW;
  int Ktot = Cin * Ktap;
  int KtotP = (Ktot + 31) & ~31;
  int cinShift = (Cin == 8) ? 3 : (Cin == 16) ? 4 : (Cin == 32) ? 5 : 6;
  TapPack tp{};
  for (int kd = 0; kd < KD; ++kd)
    for (int kh = 0; kh < KH; ++kh)
      for (int kw = 0; kw < KW; ++kw) {
        int tap = (kd * KH + kh) * KW + kw;
        tp.t[tap] = (kd << 16) | (kh << 8) | kw;
        tp.o[tap] = ((kd * in.H + kh) * in.W + kw) * Cin;
      }

  long Vout = (long)out.D * out.H * out.W;
  int  ntm  = (int)((Vout + 15) >> 4);
  int  tilesPerBlk = 16;                       // 8 waves x 2 M tiles
  dim3 grid((ntm + tilesPerBlk - 1) / tilesPerBlk, (Cout + 15) / 16);
  size_t shmem = 256 + (size_t)16 * KtotP * 2;
  hipLaunchKernelGGL(conv3d_wmma_kernel, grid, dim3(256), shmem, s,
      x, w, scale, shift, resid, y, zpad, Cin, cinShift, Cout,
      in.D, in.H, in.W, out.D, out.H, out.W,
      Ktap, Ktot, KtotP, stride, pd, ph, pw, relu, tp);
}

static void launch_tconv(hipStream_t s,
    const float* x, const float* w, const float* scale, const float* shift,
    float* y, const float* zpad, int Cin, int Cout, Dims in)
{
  int cpr = (in.W + 15) >> 4;
  long nTiles = (long)(in.D * 2) * (in.H * 2) * 2 * cpr;
  int  KP = (Cin + 31) & ~31;
  dim3 grid((unsigned)((nTiles + 7) / 8), (Cout + 15) / 16);
  size_t shmem = (size_t)8 * KP * 16 * 2;
  hipLaunchKernelGGL(tconv3d_wmma_kernel, grid, dim3(256), shmem, s,
      x, w, scale, shift, y, zpad, Cin, Cout, in.D, in.H, in.W);
}

extern "C" void kernel_launch(void* const* d_in, const int* in_sizes, int n_in,
                              void* d_out, int out_size, void* d_ws, size_t ws_size,
                              hipStream_t stream) {
  (void)in_sizes; (void)n_in; (void)out_size; (void)ws_size;

  const float* cv   = (const float*)d_in[0];
  const int*   hypo = (const int*)d_in[1];
  auto P = [&](int i) { return (const float*)d_in[i]; };

  const Dims D0{48, 160, 192}, D1{24, 80, 96}, D2{12, 40, 48}, D3{6, 20, 24};
  const long V0 = 48L * 160 * 192, V1 = 24L * 80 * 96,
             V2 = 12L * 40 * 48,  V3 = 6L * 20 * 24;

  float* ws = (float*)d_ws;
  float* X0 = ws;               // scattered input / scratch, V0*8 (channel-last)
  float* A0 = X0 + V0 * 8;
  float* B0 = A0 + V0 * 8;
  float* S0 = B0 + V0 * 8;      // conv0 skip
  float* A1 = S0 + V0 * 8;
  float* B1 = A1 + V1 * 16;
  float* S1 = B1 + V1 * 16;     // c1 skip
  float* A2 = S1 + V1 * 16;
  float* B2 = A2 + V2 * 32;
  float* S2 = B2 + V2 * 32;     // c2 skip
  float* A3 = S2 + V2 * 32;
  float* B3 = A3 + V3 * 64;
  float* C3 = B3 + V3 * 64;
  float* ZP = C3 + V3 * 64;     // 64-float zero page for OOB group loads

  // --- zero page + scatter cost volume along hypothesis depth ---
  hipMemsetAsync(ZP, 0, 64 * sizeof(float), stream);
  hipMemsetAsync(X0, 0, (size_t)V0 * 8 * sizeof(float), stream);
  hipLaunchKernelGGL(scatter_kernel, dim3((unsigned)((V0 + 255) / 256)), dim3(256),
                     0, stream, cv, hypo, X0, 48, 160, 192, 8);

  // --- input: 3x (8->8, 3^3, SAME) ---
  launch_conv(stream, X0, P(58), P(56), P(57), nullptr, A0, ZP, 8, 8, D0, D0, 3,3,3, 1, 1,1,1, 1);
  launch_conv(stream, A0, P(61), P(59), P(60), nullptr, B0, ZP, 8, 8, D0, D0, 3,3,3, 1, 1,1,1, 1);
  launch_conv(stream, B0, P(64), P(62), P(63), nullptr, S0, ZP, 8, 8, D0, D0, 3,3,3, 1, 1,1,1, 1);

  // --- conv1up: 8->16 down (2^3 s2) + 2x (16->16, 3^3) ---
  launch_conv(stream, S0, P(13), P(11), P(12), nullptr, A1, ZP, 8, 16, D0, D1, 2,2,2, 2, 0,0,0, 1);
  launch_conv(stream, A1, P(16), P(14), P(15), nullptr, B1, ZP, 16, 16, D1, D1, 3,3,3, 1, 1,1,1, 1);
  launch_conv(stream, B1, P(19), P(17), P(18), nullptr, S1, ZP, 16, 16, D1, D1, 3,3,3, 1, 1,1,1, 1);

  // --- conv2up: 16->32 down + 2x (32->32) ---
  launch_conv(stream, S1, P(31), P(29), P(30), nullptr, A2, ZP, 16, 32, D1, D2, 2,2,2, 2, 0,0,0, 1);
  launch_conv(stream, A2, P(34), P(32), P(33), nullptr, B2, ZP, 32, 32, D2, D2, 3,3,3, 1, 1,1,1, 1);
  launch_conv(stream, B2, P(37), P(35), P(36), nullptr, S2, ZP, 32, 32, D2, D2, 3,3,3, 1, 1,1,1, 1);

  // --- conv3up: 32->64 down + 2x (64->64) ---
  launch_conv(stream, S2, P(49), P(47), P(48), nullptr, A3, ZP, 32, 64, D2, D3, 2,2,2, 2, 0,0,0, 1);
  launch_conv(stream, A3, P(52), P(50), P(51), nullptr, B3, ZP, 64, 64, D3, D3, 3,3,3, 1, 1,1,1, 1);
  launch_conv(stream, B3, P(55), P(53), P(54), nullptr, C3, ZP, 64, 64, D3, D3, 3,3,3, 1, 1,1,1, 1);

  // --- conv3down: tconv 64->32 + 2 convs, skip add c2 fused in last ---
  launch_tconv(stream, C3, P(40), P(38), P(39), A2, ZP, 64, 32, D3);
  launch_conv(stream, A2, P(43), P(41), P(42), nullptr, B2, ZP, 32, 32, D2, D2, 3,3,3, 1, 1,1,1, 1);
  launch_conv(stream, B2, P(46), P(44), P(45), S2,      A2, ZP, 32, 32, D2, D2, 3,3,3, 1, 1,1,1, 1);

  // --- conv2down: tconv 32->16 + 2 convs, skip add c1 fused ---
  launch_tconv(stream, A2, P(22), P(20), P(21), A1, ZP, 32, 16, D2);
  launch_conv(stream, A1, P(25), P(23), P(24), nullptr, B1, ZP, 16, 16, D1, D1, 3,3,3, 1, 1,1,1, 1);
  launch_conv(stream, B1, P(28), P(26), P(27), S1,      A1, ZP, 16, 16, D1, D1, 3,3,3, 1, 1,1,1, 1);

  // --- conv1down: tconv 16->8 + 2 convs, skip add conv0 fused ---
  launch_tconv(stream, A1, P(4), P(2), P(3), A0, ZP, 16, 8, D1);
  launch_conv(stream, A0, P(7),  P(5), P(6), nullptr, B0, ZP, 8, 8, D0, D0, 3,3,3, 1, 1,1,1, 1);
  launch_conv(stream, B0, P(10), P(8), P(9), S0,      A0, ZP, 8, 8, D0, D0, 3,3,3, 1, 1,1,1, 1);

  // --- prob blocks: 2x (8->8, 3^3) ---
  launch_conv(stream, A0, P(67), P(65), P(66), nullptr, B0, ZP, 8, 8, D0, D0, 3,3,3, 1, 1,1,1, 1);
  launch_conv(stream, B0, P(70), P(68), P(69), nullptr, X0, ZP, 8, 8, D0, D0, 3,3,3, 1, 1,1,1, 1);

  // --- prob tail: separable raw convs, no affine/relu ---
  launch_conv(stream, X0, P(71), nullptr, nullptr, nullptr, A0, ZP, 8, 8, D0, D0, 3,1,1, 1, 1,0,0, 0);
  launch_conv(stream, A0, P(72), nullptr, nullptr, nullptr, B0, ZP, 8, 8, D0, D0, 1,1,3, 1, 0,0,1, 0);
  launch_conv(stream, B0, P(73), nullptr, nullptr, nullptr, X0, ZP, 8, 8, D0, D0, 1,3,1, 1, 0,1,0, 0);
  launch_conv(stream, X0, P(74), nullptr, nullptr, nullptr, A0, ZP, 8, 1, D0, D0, 1,1,1, 1, 0,0,0, 0);

  // --- final hypothesis gather ---
  hipLaunchKernelGGL(gather_kernel, dim3((unsigned)((V0 + 255) / 256)), dim3(256),
                     0, stream, A0, hypo, (float*)d_out, 48, 160, 192);
}
